// GraphSAGE_35931696398727
// MI455X (gfx1250) — matrix operand order repaired and checked by reading, not compiled, not run on hardware
//
#include <hip/hip_runtime.h>

#define N_NODES 40000
#define DEG     16
#define IN_F    128
#define HID_F   256
#define OUT_F   47
#define BN_EPS  1e-5f

typedef __bf16 bf16_t;
typedef __attribute__((ext_vector_type(16))) __bf16 v16bf;
typedef __attribute__((ext_vector_type(8)))  __bf16 v8bf;
typedef __attribute__((ext_vector_type(8)))  float  v8f;

// ---------------- device helpers ----------------

static __device__ __forceinline__ v8f splat8(float x) {
  v8f r;
#pragma unroll
  for (int i = 0; i < 8; ++i) r[i] = x;
  return r;
}

static __device__ __forceinline__ v16bf zero16() {
  v16bf r;
#pragma unroll
  for (int i = 0; i < 16; ++i) r[i] = (bf16_t)0.0f;
  return r;
}

// A-fragment (16x32 bf16, row-major source). Lane L: row = L%16.
// halfK = L/16. VGPR halves 0..7 = K[k0 + halfK*16 .. +8), halves 8..15 = K[k0+halfK*16+8 .. +8).
static __device__ __forceinline__ v16bf load_a_frag(const bf16_t* __restrict__ rowp,
                                                    int k0, int halfK) {
  v8bf lo = *(const v8bf*)(rowp + k0 + halfK * 16);
  v8bf hi = *(const v8bf*)(rowp + k0 + halfK * 16 + 8);
  return __builtin_shufflevector(lo, hi, 0, 1, 2, 3, 4, 5, 6, 7,
                                 8, 9, 10, 11, 12, 13, 14, 15);
}

static __device__ __forceinline__ v8f wmma_bf16(v16bf a, v16bf b, v8f c) {
  return __builtin_amdgcn_wmma_f32_16x16x32_bf16(false, a, false, b,
                                                 (short)0, c, false, false);
}

static __device__ __forceinline__ float sigmoidf(float x) {
  return 1.0f / (1.0f + __expf(-x));
}

// ---------------- utility kernels ----------------

__global__ void f32_to_bf16_k(const float* __restrict__ s, bf16_t* __restrict__ d, long n) {
  long i = (long)blockIdx.x * blockDim.x + threadIdx.x;
  long stride = (long)gridDim.x * blockDim.x;
  for (; i < n; i += stride) d[i] = (bf16_t)s[i];
}

__global__ void zero_bf16_k(bf16_t* __restrict__ p, long n) {
  long i = (long)blockIdx.x * blockDim.x + threadIdx.x;
  long stride = (long)gridDim.x * blockDim.x;
  for (; i < n; i += stride) p[i] = (bf16_t)0.0f;
}

__global__ void zero_f32_k(float* __restrict__ p, long n) {
  long i = (long)blockIdx.x * blockDim.x + threadIdx.x;
  long stride = (long)gridDim.x * blockDim.x;
  for (; i < n; i += stride) p[i] = 0.0f;
}

__global__ void bias_sum_k(const float* __restrict__ a, const float* __restrict__ b,
                           float* __restrict__ o, int n) {
  int i = blockIdx.x * blockDim.x + threadIdx.x;
  if (i < n) o[i] = a[i] + b[i];
}

// ---------------- LSTM step (one timestep) ----------------
// Grid: N/(16*ROWT) blocks x 256 threads (8 waves). Block owns ROWT*16 node rows.
// Wave w owns FI/8 columns of EACH of the 4 gates -> gate math is wave-local.
// ROWT row-tiles per wave: each B-fragment pair feeds 2*ROWT wmmas (amortizes
// the L2 weight-load latency and halves weight traffic vs ROWT=1).
template <int FI, int ROWT>
__global__ void lstm_step_kernel(const bf16_t* __restrict__ xb,   // [N, FI] bf16 (layer input)
                                 const int*    __restrict__ nbr,  // [N, DEG]
                                 const bf16_t* __restrict__ wih,  // [4FI, FI] bf16 row-major
                                 const bf16_t* __restrict__ whh,  // [4FI, FI]
                                 const float*  __restrict__ bsum, // [4FI] = b_ih + b_hh
                                 bf16_t*       __restrict__ h,    // [N, FI] bf16 state
                                 float*        __restrict__ c,    // [N, FI] f32 state
                                 int t) {
  constexpr int TPG = FI / 128;  // 16-col tiles per gate per wave
  constexpr int CPW = FI / 8;    // columns per wave (per gate)
  const int lane  = threadIdx.x & 31;
  const int wave  = threadIdx.x >> 5;
  const int l16   = lane & 15;
  const int halfK = lane >> 4;
  const int c0    = wave * CPW;

  v8f acc[4][TPG][ROWT];
#pragma unroll
  for (int g = 0; g < 4; ++g)
#pragma unroll
    for (int tn = 0; tn < TPG; ++tn) {
      v8f b = splat8(bsum[g * FI + c0 + tn * 16 + l16]);
#pragma unroll
      for (int r = 0; r < ROWT; ++r) acc[g][tn][r] = b;
    }

  const bf16_t* mrow[ROWT];
  const bf16_t* hrow[ROWT];
#pragma unroll
  for (int r = 0; r < ROWT; ++r) {
    const int row = (blockIdx.x * ROWT + r) * 16 + l16;
    const int src = nbr[row * DEG + t];
    mrow[r] = xb + (long)src * FI;
    hrow[r] = h + (long)row * FI;
  }

  for (int k0 = 0; k0 < FI; k0 += 32) {
    v16bf am[ROWT], ah[ROWT];
#pragma unroll
    for (int r = 0; r < ROWT; ++r) {
      am[r] = load_a_frag(mrow[r], k0, halfK);
      ah[r] = load_a_frag(hrow[r], k0, halfK);
    }
#pragma unroll
    for (int g = 0; g < 4; ++g) {
#pragma unroll
      for (int tn = 0; tn < TPG; ++tn) {
        const long wr = (long)(g * FI + c0 + tn * 16 + l16);
        v16bf bih = *(const v16bf*)(wih + wr * FI + k0 + halfK * 16);
        v16bf bhh = *(const v16bf*)(whh + wr * FI + k0 + halfK * 16);
#pragma unroll
        for (int r = 0; r < ROWT; ++r) {
          acc[g][tn][r] = wmma_bf16(am[r], bih, acc[g][tn][r]);
          acc[g][tn][r] = wmma_bf16(ah[r], bhh, acc[g][tn][r]);
        }
      }
    }
  }

  __syncthreads();  // all waves finished reading h rows before we overwrite them

#pragma unroll
  for (int r = 0; r < ROWT; ++r) {
#pragma unroll
    for (int tn = 0; tn < TPG; ++tn) {
      const int col = c0 + tn * 16 + l16;
#pragma unroll
      for (int v = 0; v < 8; ++v) {
        const int rr = (blockIdx.x * ROWT + r) * 16 + v + halfK * 8;  // C/D: M=v+(lane/16)*8
        const long idx = (long)rr * FI + col;
        float iv = sigmoidf(acc[0][tn][r][v]);
        float fv = sigmoidf(acc[1][tn][r][v]);
        float gv = tanhf(acc[2][tn][r][v]);
        float ov = sigmoidf(acc[3][tn][r][v]);
        float cn = fv * c[idx] + iv * gv;
        c[idx] = cn;
        h[idx] = (bf16_t)(ov * tanhf(cn));
      }
    }
  }
}

// ---------------- projection: out = x @ Wself^T + h @ Wneigh^T ----------------
// One 16x16 tile per wave; column guard handles FO=47.
template <int FI>
__global__ void proj_kernel(const bf16_t* __restrict__ xb,
                            const bf16_t* __restrict__ h,
                            const bf16_t* __restrict__ wself,   // [FO, FI]
                            const bf16_t* __restrict__ wneigh,  // [FO, FI]
                            float* __restrict__ out,            // [N, fo]
                            int fo, int colTiles) {
  const int lane   = threadIdx.x & 31;
  const int waveId = blockIdx.x * 8 + (threadIdx.x >> 5);
  const int rt = waveId / colTiles;
  const int ct = waveId % colTiles;
  if (rt >= N_NODES / 16) return;  // whole-wave uniform exit
  const int halfK = lane >> 4;
  const int l16   = lane & 15;
  const int row   = rt * 16 + l16;
  const int wr    = ct * 16 + l16;

  const bf16_t* xrow = xb + (long)row * FI;
  const bf16_t* hrow = h + (long)row * FI;

  v8f acc = splat8(0.0f);
  for (int k0 = 0; k0 < FI; k0 += 32) {
    v16bf ax = load_a_frag(xrow, k0, halfK);
    v16bf ah = load_a_frag(hrow, k0, halfK);
    v16bf bs, bn;
    if (wr < fo) {
      bs = *(const v16bf*)(wself  + (long)wr * FI + k0 + halfK * 16);
      bn = *(const v16bf*)(wneigh + (long)wr * FI + k0 + halfK * 16);
    } else {
      bs = zero16();
      bn = zero16();
    }
    acc = wmma_bf16(ax, bs, acc);
    acc = wmma_bf16(ah, bn, acc);
  }
#pragma unroll
  for (int v = 0; v < 8; ++v) {
    const int r   = rt * 16 + v + halfK * 8;
    const int col = ct * 16 + l16;
    if (col < fo) out[(long)r * fo + col] = acc[v];
  }
}

// ---------------- BatchNorm (training-mode batch stats) ----------------

__global__ void bn_stats_k(const float* __restrict__ z, float* __restrict__ stat) {
  __shared__ float s1[256];
  __shared__ float s2[256];
  const int col = blockIdx.x;
  const int tid = threadIdx.x;
  float a = 0.0f, b = 0.0f;
  for (int r = tid; r < N_NODES; r += 256) {
    float v = z[(long)r * HID_F + col];
    a += v;
    b += v * v;
  }
  s1[tid] = a; s2[tid] = b;
  __syncthreads();
  for (int s = 128; s > 0; s >>= 1) {
    if (tid < s) { s1[tid] += s1[tid + s]; s2[tid] += s2[tid + s]; }
    __syncthreads();
  }
  if (tid == 0) {
    float mu  = s1[0] / (float)N_NODES;
    float var = s2[0] / (float)N_NODES - mu * mu;  // biased variance
    stat[col]         = mu;
    stat[HID_F + col] = rsqrtf(var + BN_EPS);
  }
}

__global__ void bn_apply_k(const float* __restrict__ z,
                           const float* __restrict__ g, const float* __restrict__ b,
                           const float* __restrict__ stat,
                           bf16_t* __restrict__ out, long n) {
  long i = (long)blockIdx.x * blockDim.x + threadIdx.x;
  long stride = (long)gridDim.x * blockDim.x;
  for (; i < n; i += stride) {
    int col = (int)(i & (HID_F - 1));
    float y = g[col] * (z[i] - stat[col]) * stat[HID_F + col] + b[col];
    out[i] = (bf16_t)fmaxf(y, 0.0f);  // ReLU, to bf16 for next layer
  }
}

// ---------------- final log-softmax over 47 classes ----------------

__global__ void log_softmax_k(const float* __restrict__ z, float* __restrict__ out) {
  int row = blockIdx.x * blockDim.x + threadIdx.x;
  if (row >= N_NODES) return;
  const float* zr = z + (long)row * OUT_F;
  float m = -3.4e38f;
  for (int j = 0; j < OUT_F; ++j) m = fmaxf(m, zr[j]);
  float s = 0.0f;
  for (int j = 0; j < OUT_F; ++j) s += __expf(zr[j] - m);
  float ls = m + __logf(s);
  float* orow = out + (long)row * OUT_F;
  for (int j = 0; j < OUT_F; ++j) orow[j] = zr[j] - ls;
}

// ---------------- host orchestration ----------------

extern "C" void kernel_launch(void* const* d_in, const int* in_sizes, int n_in,
                              void* d_out, int out_size, void* d_ws, size_t ws_size,
                              hipStream_t stream) {
  (void)in_sizes; (void)n_in; (void)out_size; (void)ws_size;

  const float* x   = (const float*)d_in[0];
  const int*   nbr = (const int*)d_in[1];
  const float* W_ih[3]    = {(const float*)d_in[2],  (const float*)d_in[8],  (const float*)d_in[14]};
  const float* W_hh[3]    = {(const float*)d_in[3],  (const float*)d_in[9],  (const float*)d_in[15]};
  const float* b_ih[3]    = {(const float*)d_in[4],  (const float*)d_in[10], (const float*)d_in[16]};
  const float* b_hh[3]    = {(const float*)d_in[5],  (const float*)d_in[11], (const float*)d_in[17]};
  const float* W_self[3]  = {(const float*)d_in[6],  (const float*)d_in[12], (const float*)d_in[18]};
  const float* W_neigh[3] = {(const float*)d_in[7],  (const float*)d_in[13], (const float*)d_in[19]};
  const float* g_bn[2]    = {(const float*)d_in[20], (const float*)d_in[22]};
  const float* b_bn[2]    = {(const float*)d_in[21], (const float*)d_in[23]};

  // workspace carve-out (256B aligned)
  char* p = (char*)d_ws;
  auto alloc = [&](size_t bytes) -> char* {
    char* r = p;
    p += (bytes + 255) & ~(size_t)255;
    return r;
  };
  bf16_t* xb     = (bf16_t*)alloc((size_t)N_NODES * HID_F * 2);  // layer input, bf16
  bf16_t* hbuf   = (bf16_t*)alloc((size_t)N_NODES * HID_F * 2);  // LSTM hidden, bf16
  float*  cbuf   = (float*) alloc((size_t)N_NODES * HID_F * 4);  // LSTM cell, f32
  float*  proj   = (float*) alloc((size_t)N_NODES * HID_F * 4);  // layer output, f32
  bf16_t* wih_b  = (bf16_t*)alloc((size_t)4 * HID_F * HID_F * 2);
  bf16_t* whh_b  = (bf16_t*)alloc((size_t)4 * HID_F * HID_F * 2);
  bf16_t* wsf_b  = (bf16_t*)alloc((size_t)HID_F * HID_F * 2);
  bf16_t* wng_b  = (bf16_t*)alloc((size_t)HID_F * HID_F * 2);
  float*  bsum   = (float*) alloc((size_t)4 * HID_F * 4);
  float*  stat   = (float*) alloc((size_t)2 * HID_F * 4);

  const int rowTiles = N_NODES / 16;   // 2500
  constexpr int ROWT = 2;              // row-tiles per block in LSTM
  const int lstmBlocks = N_NODES / (16 * ROWT);  // 1250

  // ---------- layer 0 (FI=128, FO=256) ----------
  {
    const int FI = IN_F, FO = HID_F;
    f32_to_bf16_k<<<512, 256, 0, stream>>>(x, xb, (long)N_NODES * FI);
    f32_to_bf16_k<<<512, 256, 0, stream>>>(W_ih[0],  wih_b, (long)4 * FI * FI);
    f32_to_bf16_k<<<512, 256, 0, stream>>>(W_hh[0],  whh_b, (long)4 * FI * FI);
    f32_to_bf16_k<<<512, 256, 0, stream>>>(W_self[0], wsf_b, (long)FO * FI);
    f32_to_bf16_k<<<512, 256, 0, stream>>>(W_neigh[0], wng_b, (long)FO * FI);
    bias_sum_k<<<2, 256, 0, stream>>>(b_ih[0], b_hh[0], bsum, 4 * FI);
    zero_bf16_k<<<512, 256, 0, stream>>>(hbuf, (long)N_NODES * FI);
    zero_f32_k<<<512, 256, 0, stream>>>(cbuf, (long)N_NODES * FI);
    for (int t = 0; t < DEG; ++t)
      lstm_step_kernel<IN_F, ROWT><<<lstmBlocks, 256, 0, stream>>>(
          xb, nbr, wih_b, whh_b, bsum, hbuf, cbuf, t);
    const int colTiles = FO / 16;
    proj_kernel<IN_F><<<(rowTiles * colTiles + 7) / 8, 256, 0, stream>>>(
        xb, hbuf, wsf_b, wng_b, proj, FO, colTiles);
    bn_stats_k<<<HID_F, 256, 0, stream>>>(proj, stat);
    bn_apply_k<<<1024, 256, 0, stream>>>(proj, g_bn[0], b_bn[0], stat, xb,
                                         (long)N_NODES * HID_F);
  }

  // ---------- layer 1 (FI=256, FO=256) ----------
  {
    const int FI = HID_F, FO = HID_F;
    f32_to_bf16_k<<<1024, 256, 0, stream>>>(W_ih[1],  wih_b, (long)4 * FI * FI);
    f32_to_bf16_k<<<1024, 256, 0, stream>>>(W_hh[1],  whh_b, (long)4 * FI * FI);
    f32_to_bf16_k<<<512, 256, 0, stream>>>(W_self[1], wsf_b, (long)FO * FI);
    f32_to_bf16_k<<<512, 256, 0, stream>>>(W_neigh[1], wng_b, (long)FO * FI);
    bias_sum_k<<<4, 256, 0, stream>>>(b_ih[1], b_hh[1], bsum, 4 * FI);
    zero_bf16_k<<<512, 256, 0, stream>>>(hbuf, (long)N_NODES * FI);
    zero_f32_k<<<512, 256, 0, stream>>>(cbuf, (long)N_NODES * FI);
    for (int t = 0; t < DEG; ++t)
      lstm_step_kernel<HID_F, ROWT><<<lstmBlocks, 256, 0, stream>>>(
          xb, nbr, wih_b, whh_b, bsum, hbuf, cbuf, t);
    const int colTiles = FO / 16;
    proj_kernel<HID_F><<<(rowTiles * colTiles + 7) / 8, 256, 0, stream>>>(
        xb, hbuf, wsf_b, wng_b, proj, FO, colTiles);
    bn_stats_k<<<HID_F, 256, 0, stream>>>(proj, stat);
    bn_apply_k<<<1024, 256, 0, stream>>>(proj, g_bn[1], b_bn[1], stat, xb,
                                         (long)N_NODES * HID_F);
  }

  // ---------- layer 2 (FI=256, FO=47) ----------
  {
    const int FI = HID_F, FO = OUT_F;
    f32_to_bf16_k<<<1024, 256, 0, stream>>>(W_ih[2],  wih_b, (long)4 * FI * FI);
    f32_to_bf16_k<<<1024, 256, 0, stream>>>(W_hh[2],  whh_b, (long)4 * FI * FI);
    f32_to_bf16_k<<<128, 256, 0, stream>>>(W_self[2], wsf_b, (long)FO * FI);
    f32_to_bf16_k<<<128, 256, 0, stream>>>(W_neigh[2], wng_b, (long)FO * FI);
    bias_sum_k<<<4, 256, 0, stream>>>(b_ih[2], b_hh[2], bsum, 4 * FI);
    zero_bf16_k<<<512, 256, 0, stream>>>(hbuf, (long)N_NODES * FI);
    zero_f32_k<<<512, 256, 0, stream>>>(cbuf, (long)N_NODES * FI);
    for (int t = 0; t < DEG; ++t)
      lstm_step_kernel<HID_F, ROWT><<<lstmBlocks, 256, 0, stream>>>(
          xb, nbr, wih_b, whh_b, bsum, hbuf, cbuf, t);
    const int colTiles = (FO + 15) / 16;  // 3
    proj_kernel<HID_F><<<(rowTiles * colTiles + 7) / 8, 256, 0, stream>>>(
        xb, hbuf, wsf_b, wng_b, proj, FO, colTiles);
    log_softmax_k<<<(N_NODES + 255) / 256, 256, 0, stream>>>(proj, (float*)d_out);
  }
}
